// Conv1dSubsampling_1614907703436
// MI455X (gfx1250) — 8-bit matrix operand order repaired and checked by reading, compiled, not run on hardware
//
#include <hip/hip_runtime.h>
#include <stdint.h>

typedef __attribute__((ext_vector_type(8))) int v8i;

#define B_     16
#define CIN    80
#define T_     8192
#define COUT   512
#define SFX    4
#define KW     8
#define TOUT   2047
#define QK     128
#define YSTR   276
#define EPSQ   1e-5f
#define GAMMAQ 0.9f
#define QMAXQ  127.0f

#define WS_QXT  0
#define WS_QW1  (WS_QXT + (size_t)B_*T_*QK)
#define WS_SW1  (WS_QW1 + COUT*QK)
#define WS_W2Q  (WS_SW1 + COUT*4)

__global__ void quant_weights(const float* __restrict__ w1, const float* __restrict__ w2,
                              int8_t* __restrict__ qw1, float* __restrict__ sw1,
                              float* __restrict__ w2q) {
  int o = blockIdx.x * blockDim.x + threadIdx.x;
  if (o >= COUT) return;
  float m = 0.f;
  for (int c = 0; c < CIN; ++c) m = fmaxf(m, fabsf(w1[o*CIN + c]));
  float s   = fmaxf(GAMMAQ * m, EPSQ);
  float st  = s / QMAXQ;
  float inv = QMAXQ / s;
  sw1[o] = st;
  for (int c = 0; c < QK; ++c) {
    int8_t q = 0;
    if (c < CIN) {
      float wc = fminf(fmaxf(w1[o*CIN + c], -s), s);
      q = (int8_t)(int)rintf(wc * inv);
    }
    qw1[o*QK + c] = q;
  }
  float m2 = 0.f;
  for (int k = 0; k < KW; ++k) m2 = fmaxf(m2, fabsf(w2[o*KW + k]));
  float s2   = fmaxf(GAMMAQ * m2, EPSQ);
  float st2  = s2 / QMAXQ;
  float inv2 = QMAXQ / s2;
  for (int k = 0; k < KW; ++k) {
    float wc = fminf(fmaxf(w2[o*KW + k], -s2), s2);
    w2q[o*KW + k] = rintf(wc * inv2) * st2;
  }
}

#define TT 128
__launch_bounds__(256)
__global__ void quant_x_t(const float* __restrict__ x, const float* __restrict__ s1p,
                          int8_t* __restrict__ qxT) {
  __shared__ __align__(16) int8_t tle[TT * QK];
  const int b  = blockIdx.y;
  const int t0 = blockIdx.x * TT;
  const int tid = threadIdx.x;
  const float s   = fmaxf(*s1p, EPSQ);
  const float inv = QMAXQ / s;
  const float* xb = x + (size_t)b * CIN * T_;
  for (int idx = tid; idx < CIN * TT; idx += 256) {
    int k = idx / TT, tl = idx % TT;
    float v  = xb[(size_t)k * T_ + t0 + tl];
    float vc = fminf(fmaxf(v, -s), s);
    tle[tl * QK + k] = (int8_t)(int)rintf(vc * inv);
  }
  for (int idx = tid; idx < (QK - CIN) * TT; idx += 256) {
    int tl = idx % TT, k = CIN + idx / TT;
    tle[tl * QK + k] = 0;
  }
  __syncthreads();
  int* dst = (int*)(qxT + ((size_t)b * T_ + t0) * QK);
  const int* srci = (const int*)tle;
  for (int idx = tid; idx < TT * QK / 4; idx += 256)
    dst[idx] = srci[idx];
}

__global__ void finish_len(const float* __restrict__ xlen, float* __restrict__ out_tail) {
  int i = threadIdx.x;
  if (i < B_) out_tail[i] = floorf(xlen[i] / (float)SFX);
}

__launch_bounds__(128)
__global__ void fused_gemm_dw(const int8_t* __restrict__ qxT, const int8_t* __restrict__ qw1,
                              const float* __restrict__ sw1, const float* __restrict__ w2q,
                              const float* __restrict__ s1p, const float* __restrict__ s2p,
                              float* __restrict__ out) {
  __shared__ __align__(16) int8_t xT[272 * QK];
  __shared__ __align__(16) int8_t wA[16 * QK];
  __shared__ __align__(16) float  yq[16 * YSTR];

  const int tid  = threadIdx.x;
  const int lane = tid & 31;
  const int wave = tid >> 5;
  const int tt = blockIdx.x, ot = blockIdx.y, b = blockIdx.z;
  const int tbase = tt * 256;
  const int obase = ot * 16;

  const float s1    = fmaxf(*s1p, EPSQ);
  const float stepx = s1 / QMAXQ;
  const float s2    = fmaxf(*s2p, EPSQ);
  const float step2 = s2 / QMAXQ;
  const float inv2  = QMAXQ / s2;

  const int8_t* srcT = qxT + ((size_t)b * T_ + tbase) * QK;
  for (int idx = tid; idx < 272 * 8; idx += 128) {
    int row = idx >> 3, chunk = idx & 7;
    int off = row * QK + chunk * 16;
    unsigned lds = (unsigned)(uintptr_t)&xT[off];
    if (tbase + row < T_) {
      const int8_t* g = srcT + off;
      asm volatile("global_load_async_to_lds_b128 %0, %1, off"
                   :: "v"(lds), "v"(g) : "memory");
    } else {
      int4 z = {0, 0, 0, 0};
      *(int4*)&xT[off] = z;
    }
  }
  for (int idx = tid; idx < (16 * QK) / 16; idx += 128)
    *(int4*)&wA[idx * 16] = *(const int4*)&qw1[(size_t)obase * QK + idx * 16];
  asm volatile("s_wait_asynccnt 0x0" ::: "memory");
  __syncthreads();

  const int m   = lane & 15;
  const int hiA = (lane >= 16) ? 8 : 0;
  const int hiB = (lane >= 16) ? 16 : 0;
  v8i a0, a1;
#pragma unroll
  for (int r = 0; r < 8; ++r) {
    int k = ((r >> 1) << 4) + hiA + ((r & 1) << 2);
    a0[r] = *(const int*)&wA[m * QK + k];
    a1[r] = *(const int*)&wA[m * QK + 64 + k];
  }

  float scl[8];
#pragma unroll
  for (int r = 0; r < 8; ++r) scl[r] = stepx * sw1[obase + r + hiA];

  auto loadB = [&](int nt, v8i& f0, v8i& f1) {
    const int col = nt * 16 + (lane & 15);
#pragma unroll
    for (int r = 0; r < 8; ++r) {
      int k = ((r >> 1) << 4) + hiA + ((r & 1) << 2);
      f0[r] = *(const int*)&xT[col * QK + k];
      f1[r] = *(const int*)&xT[col * QK + 64 + k];
    }
  };
  auto requant = [&](const v8i& acc, int nt) {
    const int col = nt * 16 + (lane & 15);
#pragma unroll
    for (int r = 0; r < 8; ++r) {
      float y  = scl[r] * (float)acc[r];
      float yc = fminf(fmaxf(y, -s2), s2);
      yq[(r + hiA) * YSTR + col] = rintf(yc * inv2) * step2;
    }
  };

  const int nt0 = wave * 4;
  v8i f0[4], f1[4], acc[4];
#pragma unroll
  for (int i = 0; i < 4; ++i) loadB(nt0 + i, f0[i], f1[i]);
#pragma unroll
  for (int i = 0; i < 4; ++i) {
    v8i z = {};
    acc[i] = __builtin_amdgcn_wmma_i32_16x16x64_iu8(true, a0, true, f0[i], z, false, false);
  }
#pragma unroll
  for (int i = 0; i < 4; ++i)
    acc[i] = __builtin_amdgcn_wmma_i32_16x16x64_iu8(true, a1, true, f1[i], acc[i], false, false);
#pragma unroll
  for (int i = 0; i < 4; ++i) requant(acc[i], nt0 + i);

  if (wave == 0) {
    v8i g0, g1;
    loadB(16, g0, g1);
    v8i accT = {};
    accT = __builtin_amdgcn_wmma_i32_16x16x64_iu8(true, a0, true, g0, accT, false, false);
    accT = __builtin_amdgcn_wmma_i32_16x16x64_iu8(true, a1, true, g1, accT, false, false);
    requant(accT, 16);
  }
  __syncthreads();

  for (int idx = tid; idx < 16 * 64; idx += 128) {
    int mm = idx >> 6, tl = idx & 63;
    int tp = tt * 64 + tl;
    if (tp >= TOUT) continue;
    int o = obase + mm;
    const float* wk = &w2q[o * KW];
    const float* yr = &yq[mm * YSTR + tl * 4];
    float acc2 = 0.f;
#pragma unroll
    for (int k = 0; k < KW; ++k) acc2 += wk[k] * yr[k];
    out[((size_t)(b * COUT + o)) * TOUT + tp] = acc2;
  }
}

extern "C" void kernel_launch(void* const* d_in, const int* in_sizes, int n_in,
                              void* d_out, int out_size, void* d_ws, size_t ws_size,
                              hipStream_t stream) {
  const float* x    = (const float*)d_in[0];
  const float* xlen = (const float*)d_in[1];
  const float* w1   = (const float*)d_in[2];
  const float* w2   = (const float*)d_in[3];
  const float* s1   = (const float*)d_in[4];
  const float* s2   = (const float*)d_in[5];
  float* out = (float*)d_out;

  char* ws = (char*)d_ws;
  int8_t* qxT = (int8_t*)(ws + WS_QXT);
  int8_t* qw1 = (int8_t*)(ws + WS_QW1);
  float*  sw1 = (float*)(ws + WS_SW1);
  float*  w2q = (float*)(ws + WS_W2Q);

  quant_weights<<<2, 256, 0, stream>>>(w1, w2, qw1, sw1, w2q);

  dim3 qgrid(T_ / TT, B_);
  quant_x_t<<<qgrid, 256, 0, stream>>>(x, s1, qxT);

  finish_len<<<1, 32, 0, stream>>>(xlen, out + (size_t)B_ * COUT * TOUT);

  dim3 grid(32  , COUT / 16, B_);
  fused_gemm_dw<<<grid, 128, 0, stream>>>(qxT, qw1, sw1, w2q, s1, s2, out);
}
